// NonLinearDipoleReadoutBlock_28939489641119
// MI455X (gfx1250) — matrix operand order repaired and checked by reading, not compiled, not run on hardware
//
#include <hip/hip_runtime.h>
#include <cstdint>

typedef __attribute__((ext_vector_type(16))) __bf16          v16bf;
typedef __attribute__((ext_vector_type(8)))  float           v8f;
typedef __attribute__((ext_vector_type(8)))  unsigned short  v8u16;
typedef __attribute__((ext_vector_type(16))) unsigned short  v16u16;

#define K_IN            256
#define ROWS_PER_BLOCK  64
#define WAVES           4
#define THREADS         (WAVES * 32)
#define PAD             264   // row stride in elements (256 + 8) -> conflict-free LDS

__device__ __forceinline__ unsigned short f32_to_bf16(float f) {
    unsigned u = __float_as_uint(f);
    u += 0x7FFFu + ((u >> 16) & 1u);   // round-to-nearest-even
    return (unsigned short)(u >> 16);
}

__device__ __forceinline__ v16bf pack_frag(v8u16 lo, v8u16 hi) {
    v16u16 t;
#pragma unroll
    for (int i = 0; i < 8; ++i) { t[i] = lo[i]; t[8 + i] = hi[i]; }
    return __builtin_bit_cast(v16bf, t);
}

// A fragment (16x32 bf16): lanes 0-15 row M=lane hold K = {0..7, 16..23};
// lanes 16-31 same rows hold K = {8..15, 24..31}.  Two 16B LDS loads.
__device__ __forceinline__ v16bf load_a(const unsigned short* rowp, int kc, int half) {
    const unsigned short* p = rowp + kc + half * 8;
    v8u16 lo = *(const v8u16*)(p);
    v8u16 hi = *(const v8u16*)(p + 16);
    return pack_frag(lo, hi);
}

// B fragment (32x16 bf16): lane column N = lane&15; lanes 0-15 hold K = {0..7, 16..23},
// lanes 16-31 hold K = {8..15, 24..31}.  W1 staged transposed (K-contiguous).
__device__ __forceinline__ v16bf load_b(const unsigned short* wT, int col, int kc, int half) {
    const unsigned short* p = wT + col * PAD + kc + half * 8;
    v8u16 lo = *(const v8u16*)(p);
    v8u16 hi = *(const v8u16*)(p + 16);
    return pack_frag(lo, hi);
}

__device__ __forceinline__ float silu(float z) {
    return z / (1.0f + __expf(-z));
}

__global__ void __launch_bounds__(THREADS, 1)
mul_mlp_wmma_kernel(const float* __restrict__ x,
                    const float* __restrict__ w1_s,
                    const float* __restrict__ w1_v,
                    const float* __restrict__ w2_s,
                    const float* __restrict__ w2_v,
                    float* __restrict__ out,
                    int nrows)
{
    extern __shared__ unsigned short smem[];
    unsigned short* plane0 = smem;                                  // xs       64 x PAD
    unsigned short* plane1 = plane0 + ROWS_PER_BLOCK * PAD;         // xv[:,0]
    unsigned short* plane2 = plane1 + ROWS_PER_BLOCK * PAD;         // xv[:,1]
    unsigned short* plane3 = plane2 + ROWS_PER_BLOCK * PAD;         // xv[:,2]
    unsigned short* wTs    = plane3 + ROWS_PER_BLOCK * PAD;         // 128 x PAD (K-major)
    unsigned short* wTv    = wTs + 128 * PAD;                       //  64 x PAD (K-major)

    const int tid = threadIdx.x;
    const int R0  = blockIdx.x * ROWS_PER_BLOCK;

    // ---- stage W1 transposed as bf16 (K contiguous per output column) ----
    for (int idx = tid; idx < K_IN * 128; idx += THREADS) {
        int k = idx >> 7, c = idx & 127;
        wTs[c * PAD + k] = f32_to_bf16(w1_s[idx]);
    }
    for (int idx = tid; idx < K_IN * 64; idx += THREADS) {
        int k = idx >> 6, c = idx & 63;
        wTv[c * PAD + k] = f32_to_bf16(w1_v[idx]);
    }

    // ---- stage x tile: coalesced float4 reads, deinterleave 3-vector into planes ----
    for (int idx = tid; idx < ROWS_PER_BLOCK * 256; idx += THREADS) {
        int r = idx >> 8, q = idx & 255;
        int gr = R0 + r;
        float4 v = make_float4(0.f, 0.f, 0.f, 0.f);
        if (gr < nrows) v = ((const float4*)x)[(size_t)gr * 256 + q];
        float vals[4] = {v.x, v.y, v.z, v.w};
        int e0 = q * 4;
#pragma unroll
        for (int ee = 0; ee < 4; ++ee) {
            int e = e0 + ee;
            unsigned short b = f32_to_bf16(vals[ee]);
            if (e < 256) {
                plane0[r * PAD + e] = b;
            } else {
                int t = e - 256;
                int i = t % 3, mm = t / 3;
                unsigned short* pl = (i == 0) ? plane1 : (i == 1) ? plane2 : plane3;
                pl[r * PAD + mm] = b;
            }
        }
    }
    __syncthreads();

    // ---- per-wave 16-row WMMA tile ----
    const int wid  = tid >> 5;
    const int lane = tid & 31;
    const int half = lane >> 4;
    const int m    = lane & 15;
    const int rowM = wid * 16 + m;

    const unsigned short* a_s_row  = plane0 + rowM * PAD;
    const unsigned short* a_v0_row = plane1 + rowM * PAD;
    const unsigned short* a_v1_row = plane2 + rowM * PAD;
    const unsigned short* a_v2_row = plane3 + rowM * PAD;

    v8f acc_s[8];        // h_s:  16 x 128  (tiles 0..3 scalars, 4..7 gates)
    v8f acc_v[3][4];     // h_v:  16 x 64 per component
    v8f zero = {};
#pragma unroll
    for (int t = 0; t < 8; ++t) acc_s[t] = zero;
#pragma unroll
    for (int i = 0; i < 3; ++i)
#pragma unroll
        for (int t = 0; t < 4; ++t) acc_v[i][t] = zero;

#pragma unroll
    for (int kc = 0; kc < K_IN; kc += 32) {
        v16bf a_s  = load_a(a_s_row,  kc, half);
        v16bf a_v0 = load_a(a_v0_row, kc, half);
        v16bf a_v1 = load_a(a_v1_row, kc, half);
        v16bf a_v2 = load_a(a_v2_row, kc, half);

#pragma unroll
        for (int t = 0; t < 8; ++t) {
            v16bf b = load_b(wTs, 16 * t + m, kc, half);
            acc_s[t] = __builtin_amdgcn_wmma_f32_16x16x32_bf16(
                false, a_s, false, b, (short)0, acc_s[t], false, false);
        }
#pragma unroll
        for (int t = 0; t < 4; ++t) {
            v16bf b = load_b(wTv, 16 * t + m, kc, half);   // reused for 3 components
            acc_v[0][t] = __builtin_amdgcn_wmma_f32_16x16x32_bf16(
                false, a_v0, false, b, (short)0, acc_v[0][t], false, false);
            acc_v[1][t] = __builtin_amdgcn_wmma_f32_16x16x32_bf16(
                false, a_v1, false, b, (short)0, acc_v[1][t], false, false);
            acc_v[2][t] = __builtin_amdgcn_wmma_f32_16x16x32_bf16(
                false, a_v2, false, b, (short)0, acc_v[2][t], false, false);
        }
    }

    // ---- epilogue: SiLU gating + length-64 reductions, all in registers ----
    // C layout: VGPR j, lanes 0-15 -> row j, lanes 16-31 -> row j+8; column = lane&15.
    const float inv_in = 0.0625f;   // 1/sqrt(256)
    const float inv_h  = 0.125f;    // 1/sqrt(64)

#pragma unroll
    for (int j = 0; j < 8; ++j) {
        float sum_s = 0.f, sv0 = 0.f, sv1 = 0.f, sv2 = 0.f;
#pragma unroll
        for (int t = 0; t < 4; ++t) {
            int h  = 16 * t + m;
            float sc = acc_s[t][j]     * inv_in;   // scalars = h_s[:, :64]
            float gt = acc_s[t + 4][j] * inv_in;   // gates   = h_s[:, 64:]
            float gs = silu(sc);
            float gg = silu(gt);
            sum_s += gs * w2_s[h];
            float wv = gg * w2_v[h] * inv_in;      // silu(gate) * w2_v, h_v scale folded in
            sv0 += wv * acc_v[0][t][j];
            sv1 += wv * acc_v[1][t][j];
            sv2 += wv * acc_v[2][t][j];
        }
        // butterfly reduce across the 16 columns of each half-wave
#pragma unroll
        for (int off = 8; off > 0; off >>= 1) {
            sum_s += __shfl_xor(sum_s, off, 32);
            sv0   += __shfl_xor(sv0,   off, 32);
            sv1   += __shfl_xor(sv1,   off, 32);
            sv2   += __shfl_xor(sv2,   off, 32);
        }
        int gr = R0 + wid * 16 + j + 8 * half;
        if (m == 0 && gr < nrows) {
            float4 o = make_float4(sum_s * inv_h, sv0 * inv_h, sv1 * inv_h, sv2 * inv_h);
            ((float4*)out)[gr] = o;   // one contiguous 16B row per output
        }
    }
}

extern "C" void kernel_launch(void* const* d_in, const int* in_sizes, int n_in,
                              void* d_out, int out_size, void* d_ws, size_t ws_size,
                              hipStream_t stream) {
    const float* x    = (const float*)d_in[0];
    const float* w1_s = (const float*)d_in[1];
    const float* w1_v = (const float*)d_in[2];
    const float* w2_s = (const float*)d_in[3];
    const float* w2_v = (const float*)d_in[4];
    float* out = (float*)d_out;

    const int nrows   = in_sizes[0] / 1024;
    const int nblocks = (nrows + ROWS_PER_BLOCK - 1) / ROWS_PER_BLOCK;

    const size_t smem_bytes =
        (size_t)((4 * ROWS_PER_BLOCK + 128 + 64) * PAD) * sizeof(unsigned short); // ~231 KB

    (void)hipFuncSetAttribute((const void*)mul_mlp_wmma_kernel,
                              hipFuncAttributeMaxDynamicSharedMemorySize,
                              (int)smem_bytes);

    mul_mlp_wmma_kernel<<<nblocks, THREADS, smem_bytes, stream>>>(
        x, w1_s, w1_v, w2_s, w2_v, out, nrows);
}
